// DMPNN_70025146794499
// MI455X (gfx1250) — compile-verified, matrix operand order checked by reading, not run on hardware
//
#include <hip/hip_runtime.h>
#include <math.h>

// ---------------------------------------------------------------------------
// CDNA5 (gfx1250) D-MPNN forward. wave32; bf16 WMMA (f32 accum) for all large
// GEMMs; f32 everywhere else. See analysis above for roofline reasoning.
// ---------------------------------------------------------------------------

typedef __attribute__((ext_vector_type(16))) __bf16 v16bf;
typedef __attribute__((ext_vector_type(8)))  float  v8f;

union V16B { uint4 q[2]; v16bf v; };

__device__ __forceinline__ unsigned short f2bf(float f) {
  unsigned int u = __float_as_uint(f);
  u += 0x7FFFu + ((u >> 16) & 1u);          // round-to-nearest-even
  return (unsigned short)(u >> 16);
}
__device__ __forceinline__ float bf2f(unsigned short h) {
  return __uint_as_float(((unsigned int)h) << 16);
}

__device__ __forceinline__ v8f wmma_bf16(v16bf a, v16bf b, v8f c) {
  return __builtin_amdgcn_wmma_f32_16x16x32_bf16(false, a, false, b, (short)0, c,
                                                 false, false);
}

// A operand (16x32 bf16 tile) from row-major bf16 [nrows, lda].
// Lane l: M = l&15, K chunks {kb..kb+7, 16+kb..16+kb+7} with kb = (l>>4)*8
// (matches ISA 7.12.2 16-bit A layout).
__device__ __forceinline__ v16bf load_tileA(const unsigned short* __restrict__ A,
                                            long row, int lda, int k0, int lane) {
  const unsigned short* p = A + row * (long)lda + k0 + ((lane >> 4) << 3);
  V16B u;
  u.q[0] = *(const uint4*)(p);
  u.q[1] = *(const uint4*)(p + 16);
  return u.v;
}
// B operand from pre-packed buffer: each 32x16 K-tile stored as 32 lanes x
// 16 contiguous bf16 (512 elements per tile).
__device__ __forceinline__ v16bf load_tileB(const unsigned short* __restrict__ Bpk,
                                            long tile, int lane) {
  const unsigned short* p = Bpk + (tile << 9) + (lane << 4);
  V16B u;
  u.q[0] = *(const uint4*)(p);
  u.q[1] = *(const uint4*)(p + 8);
  return u.v;
}

// ---------------------------------------------------------------------------
// Weight packer: f32 [K,N] (or [N,K] if transposed) -> bf16 WMMA-B tiles.
// tile = nt*nKt + kt ; within tile: lane holds col n = nt*16 + (lane&15),
// K values {kb..kb+7, 16+kb..+7}, kb = (lane>>4)*8. Zero-pad out of range.
// ---------------------------------------------------------------------------
__global__ void pack_w_kernel(const float* __restrict__ w, int Ksrc, int Nsrc,
                              int nKt, int nNt, int transposed,
                              unsigned short* __restrict__ dst, long total) {
  long idx = (long)blockIdx.x * blockDim.x + threadIdx.x;
  if (idx >= total) return;
  int j = (int)(idx & 15);
  int lane = (int)((idx >> 4) & 31);
  long tile = idx >> 9;
  int kt = (int)(tile % nKt);
  int nt = (int)(tile / nKt);
  int n = nt * 16 + (lane & 15);
  int kb = (lane >> 4) << 3;
  int kk = (j < 8) ? (kb + j) : (16 + kb + (j - 8));
  int K = kt * 32 + kk;
  float v = 0.0f;
  if (K < Ksrc && n < Nsrc)
    v = transposed ? w[(long)n * Ksrc + K] : w[(long)K * Nsrc + n];
  dst[idx] = f2bf(v);
}

// a_bf[N,192] = bf16( concat(embed_z[x_z], x_atom) ), zero pad cols 165..191
__global__ void abf_kernel(const int* __restrict__ xz, const float* __restrict__ xa,
                           const float* __restrict__ emb, unsigned short* __restrict__ a_bf,
                           int N) {
  long idx = (long)blockIdx.x * blockDim.x + threadIdx.x;
  if (idx >= (long)N * 192) return;
  int i = (int)(idx / 192), j = (int)(idx % 192);
  float v = 0.0f;
  if (j < 32)        v = emb[xz[i] * 32 + j];
  else if (j < 165)  v = xa[(long)i * 133 + (j - 32)];
  a_bf[idx] = f2bf(v);
}

// concat_bf[N,448] = bf16( concat(a_all[165], m_to_atom[256]) ), pad 421..447
__global__ void concat_kernel(const int* __restrict__ xz, const float* __restrict__ xa,
                              const float* __restrict__ emb, const float* __restrict__ m,
                              unsigned short* __restrict__ cb, int N) {
  long idx = (long)blockIdx.x * blockDim.x + threadIdx.x;
  if (idx >= (long)N * 448) return;
  int i = (int)(idx / 448), j = (int)(idx % 448);
  float v = 0.0f;
  if (j < 32)        v = emb[xz[i] * 32 + j];
  else if (j < 165)  v = xa[(long)i * 133 + (j - 32)];
  else if (j < 421)  v = m[(long)i * 256 + (j - 165)];
  cb[idx] = f2bf(v);
}

// ---------------------------------------------------------------------------
// Generic WMMA GEMM, Nout = 256, block = 4 waves, tile 64 rows x 64 cols.
// MODE 0: outf = acc + bias                 (node hidden init)
// MODE 1: outb = bf16(relu(addend + acc))   (message GEMM, fused h_new)
// MODE 2: outf = relu(acc + bias)           (atom-out pre-LN)
// ---------------------------------------------------------------------------
template <int MODE>
__launch_bounds__(128)
__global__ void wmma_gemm_kernel(const unsigned short* __restrict__ A, int lda, long nrows,
                                 const unsigned short* __restrict__ Bpk, int nKt,
                                 const float* __restrict__ bias,
                                 const float* __restrict__ addend,
                                 float* __restrict__ outf,
                                 unsigned short* __restrict__ outb) {
  const int lane = threadIdx.x & 31;
  const int wave = threadIdx.x >> 5;
  const long r0 = (long)blockIdx.y * 64 + (long)wave * 16;
  const int c0 = blockIdx.x * 64;
  long rowA = r0 + (lane & 15);
  if (rowA >= nrows) rowA = nrows - 1;
  const long bt0 = (long)(c0 >> 4) * nKt;
  v8f acc0 = {}, acc1 = {}, acc2 = {}, acc3 = {};
  for (int kt = 0; kt < nKt; ++kt) {
    v16bf a = load_tileA(A, rowA, lda, kt * 32, lane);
    v16bf b0 = load_tileB(Bpk, bt0 + kt, lane);
    v16bf b1 = load_tileB(Bpk, bt0 + nKt + kt, lane);
    v16bf b2 = load_tileB(Bpk, bt0 + 2 * nKt + kt, lane);
    v16bf b3 = load_tileB(Bpk, bt0 + 3 * nKt + kt, lane);
    acc0 = wmma_bf16(a, b0, acc0);
    acc1 = wmma_bf16(a, b1, acc1);
    acc2 = wmma_bf16(a, b2, acc2);
    acc3 = wmma_bf16(a, b3, acc3);
  }
  const int rb = (lane >> 4) << 3;   // C layout: M = j + 8*(lane>=16)
  const int cc = lane & 15;
#pragma unroll
  for (int s = 0; s < 4; ++s) {
    v8f acc = (s == 0) ? acc0 : (s == 1) ? acc1 : (s == 2) ? acc2 : acc3;
    const int gc = c0 + s * 16 + cc;
#pragma unroll
    for (int j = 0; j < 8; ++j) {
      const long gr = r0 + rb + j;
      if (gr < nrows) {
        float v = acc[j];
        if (MODE == 0) {
          outf[gr * 256 + gc] = v + bias[gc];
        } else if (MODE == 1) {
          float x = fmaxf(addend[gr * 256 + gc] + v, 0.0f);
          outb[gr * 256 + gc] = f2bf(x);
        } else {
          outf[gr * 256 + gc] = fmaxf(v + bias[gc], 0.0f);
        }
      }
    }
  }
}

// ---------------------------------------------------------------------------
// Fused GRU: per wave, 6 accumulators (gi_{r,z,n}, gh_{r,z,n}) over K=256,
// then gate math + convex update, writing pre-LN h_e as bf16.
// grid.x = 16 col tiles, grid.y = row blocks of 64, block = 128 (4 waves).
// ---------------------------------------------------------------------------
__launch_bounds__(128)
__global__ void gru_kernel(const unsigned short* __restrict__ Hnew,
                           const unsigned short* __restrict__ Hold,
                           const unsigned short* __restrict__ Wihp,
                           const unsigned short* __restrict__ Whhp,
                           const float* __restrict__ b_ih, const float* __restrict__ b_hh,
                           const float* __restrict__ he_old,
                           unsigned short* __restrict__ hpre, long nrows) {
  const int lane = threadIdx.x & 31;
  const int wave = threadIdx.x >> 5;
  const long r0 = (long)blockIdx.y * 64 + (long)wave * 16;
  const int ct = blockIdx.x;  // 0..15 : 16-wide column tile in H=256
  long rowA = r0 + (lane & 15);
  if (rowA >= nrows) rowA = nrows - 1;
  const long t_r = (long)ct * 8, t_z = (long)(ct + 16) * 8, t_n = (long)(ct + 32) * 8;
  v8f ir = {}, iz = {}, in_ = {}, hr = {}, hz = {}, hn = {};
#pragma unroll
  for (int kt = 0; kt < 8; ++kt) {
    v16bf a1 = load_tileA(Hnew, rowA, 256, kt * 32, lane);
    v16bf a2 = load_tileA(Hold, rowA, 256, kt * 32, lane);
    ir  = wmma_bf16(a1, load_tileB(Wihp, t_r + kt, lane), ir);
    iz  = wmma_bf16(a1, load_tileB(Wihp, t_z + kt, lane), iz);
    in_ = wmma_bf16(a1, load_tileB(Wihp, t_n + kt, lane), in_);
    hr  = wmma_bf16(a2, load_tileB(Whhp, t_r + kt, lane), hr);
    hz  = wmma_bf16(a2, load_tileB(Whhp, t_z + kt, lane), hz);
    hn  = wmma_bf16(a2, load_tileB(Whhp, t_n + kt, lane), hn);
  }
  const int rb = (lane >> 4) << 3;
  const int c = ct * 16 + (lane & 15);
  const float bir = b_ih[c], biz = b_ih[256 + c], bin_ = b_ih[512 + c];
  const float bhr = b_hh[c], bhz = b_hh[256 + c], bhn = b_hh[512 + c];
#pragma unroll
  for (int j = 0; j < 8; ++j) {
    const long gr = r0 + rb + j;
    if (gr < nrows) {
      float r = 1.0f / (1.0f + expf(-(ir[j] + bir + hr[j] + bhr)));
      float z = 1.0f / (1.0f + expf(-(iz[j] + biz + hz[j] + bhz)));
      float n = tanhf(in_[j] + bin_ + r * (hn[j] + bhn));
      float ho = he_old[gr * 256 + c];
      hpre[gr * 256 + c] = f2bf((1.0f - z) * n + z * ho);
    }
  }
}

// ---------------------------------------------------------------------------
// LayerNorm over H=256 (one block per row). Optional bf16 input and output.
// var = E[x^2]-E[x]^2 (population, matches jnp.var).
// ---------------------------------------------------------------------------
__global__ void ln_kernel(const float* __restrict__ inf,
                          const unsigned short* __restrict__ inb,
                          const float* __restrict__ g, const float* __restrict__ b,
                          float* __restrict__ outf, unsigned short* __restrict__ outb) {
  const long row = blockIdx.x;
  const int t = threadIdx.x;
  float v = inb ? bf2f(inb[row * 256 + t]) : inf[row * 256 + t];
  __shared__ float s1[256], s2[256];
  s1[t] = v; s2[t] = v * v;
  __syncthreads();
  for (int o = 128; o > 0; o >>= 1) {
    if (t < o) { s1[t] += s1[t + o]; s2[t] += s2[t + o]; }
    __syncthreads();
  }
  const float mu = s1[0] * (1.0f / 256.0f);
  const float var = s2[0] * (1.0f / 256.0f) - mu * mu;
  const float y = (v - mu) * rsqrtf(var + 1e-5f) * g[t] + b[t];
  outf[row * 256 + t] = y;
  if (outb) outb[row * 256 + t] = f2bf(y);
}

// init h_e: relu(h_node[src_dir] + edge_attr@W_bond + b_bond) then LN.
// One block per directed edge (2E blocks, 256 threads).
__global__ void init_he_kernel(const float* __restrict__ h_node, const int* __restrict__ ei,
                               int E, const float* __restrict__ ea,
                               const float* __restrict__ Wb, const float* __restrict__ bb,
                               const float* __restrict__ g, const float* __restrict__ be,
                               float* __restrict__ d_he, unsigned short* __restrict__ H1) {
  const long k = blockIdx.x;
  const int t = threadIdx.x;
  const int eu = (k < E) ? (int)k : (int)(k - E);
  const int src = ei[k];  // src_dir(k) == edge_index_flat[k]
  __shared__ float attr[14];
  if (t < 14) attr[t] = ea[(long)eu * 14 + t];
  __syncthreads();
  float v = h_node[(long)src * 256 + t] + bb[t];
#pragma unroll
  for (int j = 0; j < 14; ++j) v += attr[j] * Wb[j * 256 + t];
  v = fmaxf(v, 0.0f);
  __shared__ float s1[256], s2[256];
  s1[t] = v; s2[t] = v * v;
  __syncthreads();
  for (int o = 128; o > 0; o >>= 1) {
    if (t < o) { s1[t] += s1[t + o]; s2[t] += s2[t + o]; }
    __syncthreads();
  }
  const float mu = s1[0] * (1.0f / 256.0f);
  const float var = s2[0] * (1.0f / 256.0f) - mu * mu;
  const float y = (v - mu) * rsqrtf(var + 1e-5f) * g[t] + be[t];
  d_he[k * 256 + t] = y;
  H1[k * 256 + t] = f2bf(y);
}

// segment-sum of h_e over dst_dir into m_node (4 cols/thread, f32 atomics)
__global__ void scatter_add_kernel(const float* __restrict__ he, const int* __restrict__ ei,
                                   int E, float* __restrict__ acc) {
  long idx = (long)blockIdx.x * blockDim.x + threadIdx.x;
  long k = idx >> 6;
  int c4 = (int)(idx & 63) << 2;
  if (k >= 2L * E) return;
  int dst = (k < E) ? ei[E + k] : ei[k - E];  // dst_dir
  float4 hv = *(const float4*)(he + k * 256 + c4);
  float* p = acc + (long)dst * 256 + c4;
  atomicAdd(p + 0, hv.x); atomicAdd(p + 1, hv.y);
  atomicAdd(p + 2, hv.z); atomicAdd(p + 3, hv.w);
}

// m_excl = m_node[src_dir] - h_e[rev], written directly as bf16 (8 cols/thread)
__global__ void mexcl_kernel(const float* __restrict__ m_node, const float* __restrict__ he,
                             const int* __restrict__ ei, int E,
                             unsigned short* __restrict__ Xb) {
  long idx = (long)blockIdx.x * blockDim.x + threadIdx.x;
  long k = idx >> 5;
  int c8 = (int)(idx & 31) << 3;
  if (k >= 2L * E) return;
  int s = ei[k];                               // src_dir
  long rv = (k < E) ? (k + E) : (k - E);       // rev
  const float4* mp = (const float4*)(m_node + (long)s * 256 + c8);
  const float4* hp = (const float4*)(he + rv * 256 + c8);
  float4 a0 = mp[0], a1 = mp[1], b0 = hp[0], b1 = hp[1];
  union { uint4 q; unsigned short h[8]; } o;
  o.h[0] = f2bf(a0.x - b0.x); o.h[1] = f2bf(a0.y - b0.y);
  o.h[2] = f2bf(a0.z - b0.z); o.h[3] = f2bf(a0.w - b0.w);
  o.h[4] = f2bf(a1.x - b1.x); o.h[5] = f2bf(a1.y - b1.y);
  o.h[6] = f2bf(a1.z - b1.z); o.h[7] = f2bf(a1.w - b1.w);
  *(uint4*)(Xb + k * 256 + c8) = o.q;
}

__global__ void graph_scatter_kernel(const float* __restrict__ hatom,
                                     const int* __restrict__ batch,
                                     float* __restrict__ hg, int N) {
  long idx = (long)blockIdx.x * blockDim.x + threadIdx.x;
  long k = idx >> 6;
  int c4 = (int)(idx & 63) << 2;
  if (k >= N) return;
  int bgi = batch[k];
  float4 v = *(const float4*)(hatom + k * 256 + c4);
  float* p = hg + (long)bgi * 256 + c4;
  atomicAdd(p + 0, v.x); atomicAdd(p + 1, v.y);
  atomicAdd(p + 2, v.z); atomicAdd(p + 3, v.w);
}

__global__ void readout_kernel(const float* __restrict__ hg, const float* __restrict__ Wr,
                               const float* __restrict__ br, float* __restrict__ out) {
  int idx = blockIdx.x * blockDim.x + threadIdx.x;
  if (idx >= 256 * 12) return;
  int gidx = idx / 12, t = idx % 12;
  float s = br[t];
  for (int k = 0; k < 256; ++k) s += hg[gidx * 256 + k] * Wr[k * 12 + t];
  out[idx] = s;
}

// ---------------------------------------------------------------------------
extern "C" void kernel_launch(void* const* d_in, const int* in_sizes, int n_in,
                              void* d_out, int out_size, void* d_ws, size_t ws_size,
                              hipStream_t stream) {
  const int*   x_z        = (const int*)d_in[0];
  const float* x_atom     = (const float*)d_in[1];
  const int*   ei         = (const int*)d_in[2];
  const float* edge_attr  = (const float*)d_in[3];
  const int*   batch      = (const int*)d_in[4];
  const float* embed_z    = (const float*)d_in[5];
  const float* W_atom_in  = (const float*)d_in[6];
  const float* b_atom_in  = (const float*)d_in[7];
  const float* W_bond_in  = (const float*)d_in[8];
  const float* b_bond_in  = (const float*)d_in[9];
  const float* W_msg      = (const float*)d_in[10];
  const float* W_ih       = (const float*)d_in[11];
  const float* b_ih       = (const float*)d_in[12];
  const float* W_hh       = (const float*)d_in[13];
  const float* b_hh       = (const float*)d_in[14];
  const float* W_atom_out = (const float*)d_in[15];
  const float* b_atom_out = (const float*)d_in[16];
  const float* W_read     = (const float*)d_in[17];
  const float* b_read     = (const float*)d_in[18];
  const float* g_e        = (const float*)d_in[19];
  const float* be_e       = (const float*)d_in[20];
  const float* g_a        = (const float*)d_in[21];
  const float* be_a       = (const float*)d_in[22];

  const int N = in_sizes[0];
  const int E = in_sizes[3] / 14;
  const long E2 = 2L * E;

  // outputs: (out[256*12], h_atom[N*256], h_e[2E*256]) concatenated f32
  float* out_r   = (float*)d_out;
  float* d_hatom = out_r + 256 * 12;
  float* d_he    = d_hatom + (long)N * 256;

  char* base = (char*)d_ws;
  size_t off = 0;
  auto take = [&](size_t bytes) -> char* {
    char* p = base + off;
    off += (bytes + 255) & ~(size_t)255;
    return p;
  };
  unsigned short* Wmsg_pk  = (unsigned short*)take(65536 * 2);
  unsigned short* Wih_pk   = (unsigned short*)take(196608 * 2);
  unsigned short* Whh_pk   = (unsigned short*)take(196608 * 2);
  unsigned short* Wain_pk  = (unsigned short*)take(49152 * 2);
  unsigned short* Waout_pk = (unsigned short*)take(114688 * 2);
  unsigned short* a_bf     = (unsigned short*)take((size_t)N * 192 * 2);
  float*          m_node   = (float*)take((size_t)N * 256 * 4);   // h_node then m_in_node
  unsigned short* X_bf     = (unsigned short*)take((size_t)E2 * 256 * 2);
  unsigned short* H1_bf    = (unsigned short*)take((size_t)E2 * 256 * 2);
  unsigned short* H2_bf    = (unsigned short*)take((size_t)E2 * 256 * 2);
  float*          h_graph  = (float*)take(256 * 256 * 4);
  // after the depth loop X_bf is dead -> alias concat + h_atom_pre on it
  unsigned short* concat_bf = X_bf;
  float* hatom_pre = (float*)((char*)X_bf + (((size_t)N * 448 * 2 + 255) & ~(size_t)255));

  // ---- pack weights to bf16 WMMA B-tiles (once per call; tiny) ----
  auto pk = [&](const float* w, int Ks, int Ns, int nKt, int nNt, int tr,
                unsigned short* dst) {
    long total = (long)nNt * nKt * 512;
    pack_w_kernel<<<dim3((unsigned)((total + 255) / 256)), 256, 0, stream>>>(
        w, Ks, Ns, nKt, nNt, tr, dst, total);
  };
  pk(W_msg,      256, 256,  8, 16, 0, Wmsg_pk);
  pk(W_ih,       256, 768,  8, 48, 1, Wih_pk);    // effective B = W_ih^T
  pk(W_hh,       256, 768,  8, 48, 1, Whh_pk);    // effective B = W_hh^T
  pk(W_atom_in,  165, 256,  6, 16, 0, Wain_pk);   // K padded 165 -> 192
  pk(W_atom_out, 421, 256, 14, 16, 0, Waout_pk);  // K padded 421 -> 448

  // ---- node features and initial hidden ----
  abf_kernel<<<dim3((unsigned)(((long)N * 192 + 255) / 256)), 256, 0, stream>>>(
      x_z, x_atom, embed_z, a_bf, N);
  wmma_gemm_kernel<0><<<dim3(4, (unsigned)((N + 63) / 64)), 128, 0, stream>>>(
      a_bf, 192, (long)N, Wain_pk, 6, b_atom_in, nullptr, m_node, nullptr);
  init_he_kernel<<<dim3((unsigned)E2), 256, 0, stream>>>(
      m_node, ei, E, edge_attr, W_bond_in, b_bond_in, g_e, be_e, d_he, H1_bf);

  // ---- message passing depth loop ----
  const unsigned rbE = (unsigned)((E2 + 63) / 64);
  for (int d = 0; d < 3; ++d) {
    hipMemsetAsync(m_node, 0, (size_t)N * 256 * 4, stream);
    scatter_add_kernel<<<dim3((unsigned)((E2 * 64 + 255) / 256)), 256, 0, stream>>>(
        d_he, ei, E, m_node);
    mexcl_kernel<<<dim3((unsigned)((E2 * 32 + 255) / 256)), 256, 0, stream>>>(
        m_node, d_he, ei, E, X_bf);
    // h_new = relu(h_e + m_excl @ W_msg)  -> bf16
    wmma_gemm_kernel<1><<<dim3(4, rbE), 128, 0, stream>>>(
        X_bf, 256, E2, Wmsg_pk, 8, nullptr, d_he, nullptr, H2_bf);
    // fused GRU -> pre-LN h_e (bf16, into retired X_bf)
    gru_kernel<<<dim3(16, rbE), 128, 0, stream>>>(
        H2_bf, H1_bf, Wih_pk, Whh_pk, b_ih, b_hh, d_he, X_bf, E2);
    // LN -> d_he (f32 state/output) + H1_bf (bf16 for next iteration)
    ln_kernel<<<dim3((unsigned)E2), 256, 0, stream>>>(
        nullptr, X_bf, g_e, be_e, d_he, H1_bf);
  }

  // ---- atom readout ----
  hipMemsetAsync(m_node, 0, (size_t)N * 256 * 4, stream);
  scatter_add_kernel<<<dim3((unsigned)((E2 * 64 + 255) / 256)), 256, 0, stream>>>(
      d_he, ei, E, m_node);  // m_to_atom
  concat_kernel<<<dim3((unsigned)(((long)N * 448 + 255) / 256)), 256, 0, stream>>>(
      x_z, x_atom, embed_z, m_node, concat_bf, N);
  wmma_gemm_kernel<2><<<dim3(4, (unsigned)((N + 63) / 64)), 128, 0, stream>>>(
      concat_bf, 448, (long)N, Waout_pk, 14, b_atom_out, nullptr, hatom_pre, nullptr);
  ln_kernel<<<dim3((unsigned)N), 256, 0, stream>>>(
      hatom_pre, nullptr, g_a, be_a, d_hatom, nullptr);

  // ---- graph pooling + linear readout ----
  hipMemsetAsync(h_graph, 0, 256 * 256 * 4, stream);
  graph_scatter_kernel<<<dim3((unsigned)(((long)N * 64 + 255) / 256)), 256, 0, stream>>>(
      d_hatom, batch, h_graph, N);
  readout_kernel<<<dim3(12), 256, 0, stream>>>(h_graph, W_read, b_read, out_r);
}
